// Sequence_747324310225
// MI455X (gfx1250) — compile-verified, matrix operand order checked by reading, not run on hardware
//
#include <hip/hip_runtime.h>
#include <hip/hip_fp16.h>

typedef __attribute__((ext_vector_type(16))) _Float16 v16h;
typedef __attribute__((ext_vector_type(8)))  _Float16 v8h;
typedef __attribute__((ext_vector_type(8)))  float    v8f;

#define HDIM 51
#define LOG2E 1.44269504088896340736f

__device__ __forceinline__ float fast_sigmoid(float z) {
  float e = __builtin_amdgcn_exp2f(-LOG2E * z);
  return __builtin_amdgcn_rcpf(1.0f + e);
}
__device__ __forceinline__ float fast_tanh(float z) {
  // tanh(z) = 1 - 2/(exp(2z)+1); safe at both infinities
  float e = __builtin_amdgcn_exp2f(2.0f * LOG2E * z);
  return 1.0f - 2.0f * __builtin_amdgcn_rcpf(e + 1.0f);
}

// One wave owns 16 batch rows for the whole time scan.
// Gate space padded to 256 cols: col = q*64 + j  (q = gate i/f/g/o, j = hidden unit, j<51 valid)
// B-fragments pre-swizzled to the 16-bit WMMA operand layout:
//   lane L: n = L%16, hiK = (L<16?0:8); half e -> Klocal = e<8 ? hiK+e : 16+hiK+(e-8)
__global__ __launch_bounds__(128)
void lstm2_wmma_kernel(const float* __restrict__ input,
                       const float* __restrict__ w_ih1, const float* __restrict__ w_hh1,
                       const float* __restrict__ b_ih1, const float* __restrict__ b_hh1,
                       const float* __restrict__ w_ih2, const float* __restrict__ w_hh2,
                       const float* __restrict__ b_ih2, const float* __restrict__ b_hh2,
                       const float* __restrict__ w_lin, const float* __restrict__ b_lin,
                       float* __restrict__ out, int T, int Tout)
{
  __shared__ v16h b1frag[32][32];                     // 32 KB: layer1 W (K=64  x N=256)
  __shared__ v16h b2frag[64][32];                     // 64 KB: layer2 [Wih2|Whh2] (K=128 x N=256)
  __shared__ __align__(16) _Float16 hcat[4][16][136]; // per-wave [h1(0..63) | h2(64..127)] + pad
  __shared__ float bias1s[256], bias2s[256], wih1s[256];
  __shared__ float xtile[4][16];

  const int tid = threadIdx.x;

  // ---------------- one-time weight prep into WMMA B-fragment layout ----------------
  for (int idx = tid; idx < 32 * 512; idx += 128) {
    int f = idx >> 9, rem = idx & 511, l = rem >> 4, e = rem & 15;
    int ntile = f >> 1, ks = f & 1;
    int cc = l & 15, hh = (l & 16) ? 8 : 0;
    int Kl = (e < 8) ? (hh + e) : (16 + hh + (e - 8));
    int K = ks * 32 + Kl;
    int col = ntile * 16 + cc, q = col >> 6, j = col & 63;
    float v = 0.0f;
    if (K < HDIM && j < HDIM) v = w_hh1[(q * HDIM + j) * HDIM + K];
    ((_Float16*)&b1frag[f][l])[e] = (_Float16)v;
  }
  for (int idx = tid; idx < 64 * 512; idx += 128) {
    int f = idx >> 9, rem = idx & 511, l = rem >> 4, e = rem & 15;
    int ntile = f >> 2, ks = f & 3;
    int cc = l & 15, hh = (l & 16) ? 8 : 0;
    int Kl = (e < 8) ? (hh + e) : (16 + hh + (e - 8));
    int K = ks * 32 + Kl;
    int col = ntile * 16 + cc, q = col >> 6, j = col & 63;
    float v = 0.0f;
    if (j < HDIM) {
      if (K < 64) { if (K < HDIM) v = w_ih2[(q * HDIM + j) * HDIM + K]; }
      else        { int kk = K - 64; if (kk < HDIM) v = w_hh2[(q * HDIM + j) * HDIM + kk]; }
    }
    ((_Float16*)&b2frag[f][l])[e] = (_Float16)v;
  }
  for (int col = tid; col < 256; col += 128) {
    int q = col >> 6, j = col & 63;
    float bb1 = 0.f, bb2 = 0.f, wi = 0.f;
    if (j < HDIM) {
      int r = q * HDIM + j;
      bb1 = b_ih1[r] + b_hh1[r];
      bb2 = b_ih2[r] + b_hh2[r];
      wi  = w_ih1[r];               // w_ih1 is [204,1]
    }
    bias1s[col] = bb1; bias2s[col] = bb2; wih1s[col] = wi;
  }
  for (int i = tid; i < 4 * 16 * 136; i += 128) ((_Float16*)hcat)[i] = (_Float16)0.0f;
  __syncthreads();

  // ---------------- per-wave setup ----------------
  const int w    = tid >> 5;
  const int lane = tid & 31;
  const int cix  = lane & 15;             // column within 16-tile / A-row index
  const int hi   = (lane & 16) ? 8 : 0;   // row half owned by this lane (C layout)
  const int rowbase = blockIdx.x * 64 + w * 16;

  float biasv1[16], biasv2[16], wihv[16];
#pragma unroll
  for (int n = 0; n < 16; ++n) {
    int col = n * 16 + cix;
    biasv1[n] = bias1s[col]; biasv2[n] = bias2s[col]; wihv[n] = wih1s[col];
  }
  float wlinv[4];
#pragma unroll
  for (int tt = 0; tt < 4; ++tt) {
    int col = tt * 16 + cix;
    wlinv[tt] = (col < HDIM) ? w_lin[col] : 0.0f;
  }
  const float blin = b_lin[0];

  float c1s[4][8], c2s[4][8], ylast[8];
#pragma unroll
  for (int a = 0; a < 4; ++a)
#pragma unroll
    for (int k = 0; k < 8; ++k) { c1s[a][k] = 0.f; c2s[a][k] = 0.f; }
#pragma unroll
  for (int k = 0; k < 8; ++k) ylast[k] = 0.f;

  // ---------------- time scan ----------------
  for (int t = 0; t < Tout; ++t) {
    float xv[8];
    if (t < T) {
      if (lane < 16) xtile[w][lane] = input[(long)(rowbase + lane) * T + t];
#pragma unroll
      for (int k = 0; k < 8; ++k) xv[k] = xtile[w][hi + k];   // DS in-order within wave
    } else {
#pragma unroll
      for (int k = 0; k < 8; ++k) xv[k] = ylast[k];           // future: feed y back
    }

    // ===== layer 1: gates = h1 @ W1' + x*wih1 + b1 =====
    v16h a1[2];
#pragma unroll
    for (int ks = 0; ks < 2; ++ks) {
      const _Float16* bp = &hcat[w][cix][ks * 32 + hi];
      v8h lo = *(const v8h*)bp;
      v8h hp = *(const v8h*)(bp + 16);
      a1[ks] = __builtin_shufflevector(lo, hp, 0,1,2,3,4,5,6,7,8,9,10,11,12,13,14,15);
    }
#pragma unroll
    for (int tt = 0; tt < 4; ++tt) {
      v8f acc[4];
#pragma unroll
      for (int q = 0; q < 4; ++q) {
        int nt = q * 4 + tt;
        float bb = biasv1[nt];
        v8f cacc = {bb, bb, bb, bb, bb, bb, bb, bb};
#pragma unroll
        for (int ks = 0; ks < 2; ++ks)
          cacc = __builtin_amdgcn_wmma_f32_16x16x32_f16(false, a1[ks], false,
                    b1frag[nt * 2 + ks][lane], (short)0, cacc, false, false);
        float wv = wihv[nt];
#pragma unroll
        for (int k = 0; k < 8; ++k) cacc[k] += xv[k] * wv;    // rank-1 x term
        acc[q] = cacc;
      }
      int col = tt * 16 + cix;
      bool valid = (col < HDIM);
#pragma unroll
      for (int k = 0; k < 8; ++k) {
        float ig = fast_sigmoid(acc[0][k]);
        float fg = fast_sigmoid(acc[1][k]);
        float gg = fast_tanh(acc[2][k]);
        float og = fast_sigmoid(acc[3][k]);
        float cn = fg * c1s[tt][k] + ig * gg;
        c1s[tt][k] = cn;
        float hn = og * fast_tanh(cn);
        hcat[w][hi + k][col] = valid ? (_Float16)hn : (_Float16)0.0f;
      }
    }

    // ===== layer 2: gates = [h1|h2] @ W2' + b2 =====
    v16h a2[4];
#pragma unroll
    for (int ks = 0; ks < 4; ++ks) {
      const _Float16* bp = &hcat[w][cix][ks * 32 + hi];
      v8h lo = *(const v8h*)bp;
      v8h hp = *(const v8h*)(bp + 16);
      a2[ks] = __builtin_shufflevector(lo, hp, 0,1,2,3,4,5,6,7,8,9,10,11,12,13,14,15);
    }
    float ys[8];
#pragma unroll
    for (int k = 0; k < 8; ++k) ys[k] = 0.f;
#pragma unroll
    for (int tt = 0; tt < 4; ++tt) {
      v8f acc[4];
#pragma unroll
      for (int q = 0; q < 4; ++q) {
        int nt = q * 4 + tt;
        float bb = biasv2[nt];
        v8f cacc = {bb, bb, bb, bb, bb, bb, bb, bb};
#pragma unroll
        for (int ks = 0; ks < 4; ++ks)
          cacc = __builtin_amdgcn_wmma_f32_16x16x32_f16(false, a2[ks], false,
                    b2frag[nt * 4 + ks][lane], (short)0, cacc, false, false);
        acc[q] = cacc;
      }
      int col = tt * 16 + cix;
      bool valid = (col < HDIM);
      float wl = wlinv[tt];
#pragma unroll
      for (int k = 0; k < 8; ++k) {
        float ig = fast_sigmoid(acc[0][k]);
        float fg = fast_sigmoid(acc[1][k]);
        float gg = fast_tanh(acc[2][k]);
        float og = fast_sigmoid(acc[3][k]);
        float cn = fg * c2s[tt][k] + ig * gg;
        c2s[tt][k] = cn;
        float hn = og * fast_tanh(cn);
        hcat[w][hi + k][64 + col] = valid ? (_Float16)hn : (_Float16)0.0f;
        ys[k] += (valid ? hn : 0.0f) * wl;
      }
    }

    // readout: reduce over the 16 column-lanes (stays within each half of wave32)
#pragma unroll
    for (int mask = 1; mask <= 8; mask <<= 1)
#pragma unroll
      for (int k = 0; k < 8; ++k) ys[k] += __shfl_xor(ys[k], mask, 32);
#pragma unroll
    for (int k = 0; k < 8; ++k) ylast[k] = ys[k] + blin;
    if (cix == 0) {
#pragma unroll
      for (int k = 0; k < 8; ++k)
        out[(long)(rowbase + hi + k) * Tout + t] = ylast[k];
    }
  }
}

extern "C" void kernel_launch(void* const* d_in, const int* in_sizes, int n_in,
                              void* d_out, int out_size, void* d_ws, size_t ws_size,
                              hipStream_t stream) {
  const float* input = (const float*)d_in[0];
  // d_in[1] = future (int scalar); handled via Tout = out_size / B
  const float* w_ih1 = (const float*)d_in[2];
  const float* w_hh1 = (const float*)d_in[3];
  const float* b_ih1 = (const float*)d_in[4];
  const float* b_hh1 = (const float*)d_in[5];
  const float* w_ih2 = (const float*)d_in[6];
  const float* w_hh2 = (const float*)d_in[7];
  const float* b_ih2 = (const float*)d_in[8];
  const float* b_hh2 = (const float*)d_in[9];
  const float* w_lin = (const float*)d_in[10];
  const float* b_lin = (const float*)d_in[11];
  float* out = (float*)d_out;

  const int B = 1024;                 // fixed by the reference setup
  const int T = in_sizes[0] / B;
  const int Tout = out_size / B;

  // 64 row-tiles of 16 rows: 16 blocks x 4 waves (128 threads, wave32)
  lstm2_wmma_kernel<<<dim3(B / 64), dim3(128), 0, stream>>>(
      input, w_ih1, w_hh1, b_ih1, b_hh1, w_ih2, w_hh2, b_ih2, b_hh2,
      w_lin, b_lin, out, T, Tout);
}
